// Seq2Seq_19756849561744
// MI455X (gfx1250) — matrix operand order repaired and checked by reading, not compiled, not run on hardware
//
#include <hip/hip_runtime.h>

typedef __attribute__((ext_vector_type(16))) __bf16 v16bf;
typedef __attribute__((ext_vector_type(8)))  __bf16 v8bf;
typedef __attribute__((ext_vector_type(8)))  float  v8f;

#define B_TOT 2048
#define T_LEN 128
#define F_IN  32
#define HID   64
#define NZ    256      // 4*HID
#define P_OUT 24

#define WAVES 4
#define BLOCK (WAVES * 32)     // 128 threads
#define ROWS_PER_WG 16         // all 4 waves cooperate on one 16-row tile

// LDS layout (bytes)
#define OFF_WXT 0                    // 256*32 bf16 = 16384 B  (Wx^T staging, [n][k])
#define OFF_WHT 16384                // 256*64 bf16 = 32768 B  (Wh^T staging, [n][k])
#define OFF_B   49152                // 256 f32 = 1024 B
#define OFF_H   50176                // 2 x (16*64) bf16 ping-pong h buffers (2KB each)
#define SMEM_SZ (50176 + 2 * 2048)

// ---- hardware tanh (CDNA5 lists TANH as a TRANS VALU op) ----
static __device__ __forceinline__ float tanh_hw(float x) {
#if __has_builtin(__builtin_amdgcn_tanhf)
  return __builtin_amdgcn_tanhf(x);
#elif __has_builtin(__builtin_amdgcn_tanh_f32)
  return __builtin_amdgcn_tanh_f32(x);
#else
  float y;
  // v_nop: TRANS result needs 1 independent op before use (ISA 7.4); the
  // scheduler cannot see through inline asm, so embed it.
  asm volatile("v_tanh_f32 %0, %1\n\tv_nop" : "=v"(y) : "v"(x));
  return y;
#endif
}

static __device__ __forceinline__ float sigmoid_f(float x) {
  // sigma(x) = 0.5*tanh(0.5x) + 0.5  -> mul + v_tanh_f32 + fma (no exp/rcp)
  return fmaf(0.5f, tanh_hw(0.5f * x), 0.5f);
}

static __device__ __forceinline__ v16bf cat8(v8bf lo, v8bf hi) {
  v16bf r;
#pragma unroll
  for (int i = 0; i < 8; ++i) { r[i] = lo[i]; r[i + 8] = hi[i]; }
  return r;
}

static __device__ __forceinline__ v16bf ld_frag16(const __bf16* p) {
  // 16 contiguous bf16 -> one fragment (two ds_load_b128)
  return cat8(*(const v8bf*)p, *(const v8bf*)(p + 8));
}

__global__ __launch_bounds__(BLOCK) void seq2seq_lstm_wmma(
    const float* __restrict__ x,
    const float* __restrict__ WxE, const float* __restrict__ WhE, const float* __restrict__ bE,
    const float* __restrict__ WxD, const float* __restrict__ WhD, const float* __restrict__ bD,
    const float* __restrict__ Wf1, const float* __restrict__ bfc1,
    const float* __restrict__ Wf2, const float* __restrict__ bfc2,
    float* __restrict__ out) {
  __shared__ __align__(16) unsigned char smem[SMEM_SZ];
  __bf16* sWxT  = (__bf16*)(smem + OFF_WXT);
  __bf16* sWhT  = (__bf16*)(smem + OFF_WHT);
  float*  sBias = (float*)(smem + OFF_B);
  __bf16* sH    = (__bf16*)(smem + OFF_H);   // 2 ping-pong buffers, 16x64 bf16 each

  const int tid  = threadIdx.x;
  const int lane = tid & 31;
  const int wv   = tid >> 5;            // 0..3 : hidden-column block owned by this wave
  const int half = (lane >> 4) & 1;
  const int lq   = lane & 15;

  const int wgBase = blockIdx.x * ROWS_PER_WG;

  // c-state slice in WMMA C-layout: cst[rr] <-> (row rr+8*half, hidden 16*wv+lq)
  float cst[8];
#pragma unroll
  for (int r = 0; r < 8; ++r) cst[r] = 0.0f;

  // zero the first h ping-pong buffer (h0 = 0)
  for (int i = tid; i < 16 * HID; i += BLOCK) sH[i] = (__bf16)0.0f;

  const float* WxL[2] = {WxE, WxD};
  const float* WhL[2] = {WhE, WhD};
  const float* bL[2]  = {bE, bD};

  const float* xrow = x + (size_t)(wgBase + lq) * (T_LEN * F_IN);

  int sel = 0;  // ping-pong selector, carried across encoder->decoder

  for (int L = 0; L < 2; ++L) {
    __syncthreads();
    // Convert this layer's weights f32 -> bf16, transposed [n][k] into LDS staging
    {
      const float* Wx = WxL[L];
      const float* Wh = WhL[L];
      const float* bb = bL[L];
      for (int i = tid; i < NZ * F_IN; i += BLOCK)
        sWxT[i] = (__bf16)Wx[(i & (F_IN - 1)) * NZ + (i >> 5)];
      for (int i = tid; i < NZ * HID; i += BLOCK)
        sWhT[i] = (__bf16)Wh[(i & (HID - 1)) * NZ + (i >> 6)];
      for (int i = tid; i < NZ; i += BLOCK) sBias[i] = bb[i];
    }
    __syncthreads();

    // ---- Hoist this wave's STATIONARY B-operands into registers:
    //      12 fragments (4 gate-tiles x {Wx, Wh.k0, Wh.k1}) = 96 VGPRs.
    v16bf bx[4], bh0[4], bh1[4];
    float bv[4];
#pragma unroll
    for (int g = 0; g < 4; ++g) {
      const int nt = 4 * g + wv;                    // gate g, hidden block wv
      bv[g]  = sBias[nt * 16 + lq];
      bx[g]  = cat8(*(const v8bf*)(sWxT + (nt * 16 + lq) * F_IN + half * 8), *(const v8bf*)(sWxT + (nt * 16 + lq) * F_IN + 16 + half * 8));
      bh0[g] = cat8(*(const v8bf*)(sWhT + (nt * 16 + lq) * HID + half * 8), *(const v8bf*)(sWhT + (nt * 16 + lq) * HID + 16 + half * 8));
      bh1[g] = cat8(*(const v8bf*)(sWhT + (nt * 16 + lq) * HID + 32 + half * 8), *(const v8bf*)(sWhT + (nt * 16 + lq) * HID + 48 + half * 8));
    }

    for (int t = 0; t < T_LEN; ++t) {
      __syncthreads();  // h(t-1) writes visible; prev-step reads done (ping-pong WAR safe)

      const __bf16* rd = sH + sel * (16 * HID);
      __bf16*       wr = sH + (sel ^ 1) * (16 * HID);
      sel ^= 1;

      // ---- A fragments for h(t-1)
      const __bf16* hp = rd + lq * HID + half * 8;
      v16bf ah0 = cat8(*(const v8bf*)(hp),      *(const v8bf*)(hp + 16));
      v16bf ah1 = cat8(*(const v8bf*)(hp + 32), *(const v8bf*)(hp + 48));

      // ---- A fragment for x_t (lane m=lq, K = half*8+{0..7}, 16+half*8+{0..7})
      const float* xp = xrow + t * F_IN;
      if (t + 1 < T_LEN) __builtin_prefetch(xp + F_IN, 0, 3);  // global_prefetch
      v16bf ax;
      {
        const float* p0 = xp + half * 8;
        const float4 a0 = *(const float4*)(p0);
        const float4 a1 = *(const float4*)(p0 + 4);
        const float4 a2 = *(const float4*)(p0 + 16);
        const float4 a3 = *(const float4*)(p0 + 20);
        ax[0]  = (__bf16)a0.x; ax[1]  = (__bf16)a0.y; ax[2]  = (__bf16)a0.z; ax[3]  = (__bf16)a0.w;
        ax[4]  = (__bf16)a1.x; ax[5]  = (__bf16)a1.y; ax[6]  = (__bf16)a1.z; ax[7]  = (__bf16)a1.w;
        ax[8]  = (__bf16)a2.x; ax[9]  = (__bf16)a2.y; ax[10] = (__bf16)a2.z; ax[11] = (__bf16)a2.w;
        ax[12] = (__bf16)a3.x; ax[13] = (__bf16)a3.y; ax[14] = (__bf16)a3.z; ax[15] = (__bf16)a3.w;
      }

      // ---- z = bias + x@Wx + h@Wh, A-operand-major (4 independent chains/phase)
      v8f z[4];
#pragma unroll
      for (int g = 0; g < 4; ++g) {
#pragma unroll
        for (int i = 0; i < 8; ++i) z[g][i] = bv[g];
      }
#pragma unroll
      for (int g = 0; g < 4; ++g)
        z[g] = __builtin_amdgcn_wmma_f32_16x16x32_bf16(
            false, ax, false, bx[g], (short)0, z[g], false, false);
#pragma unroll
      for (int g = 0; g < 4; ++g)
        z[g] = __builtin_amdgcn_wmma_f32_16x16x32_bf16(
            false, ah0, false, bh0[g], (short)0, z[g], false, false);
#pragma unroll
      for (int g = 0; g < 4; ++g)
        z[g] = __builtin_amdgcn_wmma_f32_16x16x32_bf16(
            false, ah1, false, bh1[g], (short)0, z[g], false, false);

      // ---- gate activations + state update (tanh-unit only, no exp/rcp)
#pragma unroll
      for (int rr = 0; rr < 8; ++rr) {
        float ig = sigmoid_f(z[0][rr]);
        float fg = sigmoid_f(z[1][rr]);
        float gg = tanh_hw(z[2][rr]);
        float og = sigmoid_f(z[3][rr]);
        float cn = fg * cst[rr] + ig * gg;
        cst[rr] = cn;
        float hn = og * tanh_hw(cn);
        wr[(rr + 8 * half) * HID + wv * 16 + lq] = (__bf16)hn;
      }
    }  // t
  }    // L (encoder, decoder)

  // ---- MLP head: pred = relu(h @ Wf1 + b1) @ Wf2 + b2
  __syncthreads();                 // final h written; weight staging LDS now dead
  const __bf16* hfin = sH + sel * (16 * HID);
  float* sRelu = (float*)smem;     // overlay on dead weight staging, 16*64 f32 = 4KB

#pragma unroll
  for (int i = 0; i < 8; ++i) {    // 16 rows x 64 cols = 1024 outputs / 128 threads
    int idx = i * BLOCK + tid;
    int r = idx >> 6, c = idx & 63;
    float acc = bfc1[c];
    for (int k = 0; k < HID; ++k)
      acc += (float)hfin[r * HID + k] * Wf1[k * HID + c];
    sRelu[r * HID + c] = fmaxf(acc, 0.0f);
  }
  __syncthreads();
#pragma unroll
  for (int i = 0; i < 3; ++i) {    // 16 rows x 24 cols = 384 outputs / 128 threads
    int idx = i * BLOCK + tid;
    int r = idx / 24, j = idx % 24;
    float acc = bfc2[j];
    for (int k = 0; k < HID; ++k)
      acc += sRelu[r * HID + k] * Wf2[k * P_OUT + j];
    out[(size_t)(wgBase + r) * P_OUT + j] = acc;
  }
}

extern "C" void kernel_launch(void* const* d_in, const int* in_sizes, int n_in,
                              void* d_out, int out_size, void* d_ws, size_t ws_size,
                              hipStream_t stream) {
  (void)in_sizes; (void)n_in; (void)d_ws; (void)ws_size; (void)out_size;
  const float* x_in = (const float*)d_in[0];
  const float* WxE  = (const float*)d_in[1];
  const float* WhE  = (const float*)d_in[2];
  const float* bE   = (const float*)d_in[3];
  const float* WxD  = (const float*)d_in[4];
  const float* WhD  = (const float*)d_in[5];
  const float* bD   = (const float*)d_in[6];
  const float* Wf1  = (const float*)d_in[7];
  const float* bf1  = (const float*)d_in[8];
  const float* Wf2  = (const float*)d_in[9];
  const float* bf2  = (const float*)d_in[10];
  float* out = (float*)d_out;

  dim3 grid(B_TOT / ROWS_PER_WG);  // 128 workgroups
  dim3 block(BLOCK);               // 4 waves each
  hipLaunchKernelGGL(seq2seq_lstm_wmma, grid, block, 0, stream,
                     x_in, WxE, WhE, bE, WxD, WhD, bD, Wf1, bf1, Wf2, bf2, out);
}
